// HGraphConvLayer_3143916060812
// MI455X (gfx1250) — matrix operand order repaired and checked by reading, not compiled, not run on hardware
//
#include <hip/hip_runtime.h>
#include <hip/hip_bf16.h>

typedef __attribute__((ext_vector_type(16))) __bf16 v16bf;
typedef __attribute__((ext_vector_type(8)))  float  v8f;

union Frag  { v16bf v; uint4 q[2]; };        // 32B: one WMMA A/B fragment per lane
union Pack8 { __bf16 h[8]; uint4 q; };       // 16B: 8 packed bf16

#define LDA 136  // padded bf16 row stride for the A/H tile (272B -> 4-bank rotation, 16B aligned)

// ---------------- utility kernels ----------------

__global__ void zero_f(float* p, long n) {
  long i = (long)blockIdx.x * blockDim.x + threadIdx.x;
  long stride = (long)gridDim.x * blockDim.x;
  for (; i < n; i += stride) p[i] = 0.0f;
}

// swizzle W (128 x N row-major f32) into bf16 WMMA-B fragment order:
// o[(((kb*(N/16) + t)*32 + lane)*16) + e] = W[(kb*32 + (lane>>4)*8 + (e&7) + (e>>3)*16)*N + (t*16 + (lane&15))]
__global__ void swz_b(const float* __restrict__ W, __bf16* __restrict__ o, int N) {
  int i = blockIdx.x * blockDim.x + threadIdx.x;
  if (i >= 128 * N) return;
  int e    = i & 15;
  int lane = (i >> 4) & 31;
  int tt   = (i >> 9) % (N >> 4);
  int kb   = (i >> 9) / (N >> 4);
  int k = kb * 32 + (lane >> 4) * 8 + (e & 7) + (e >> 3) * 16;
  int n = tt * 16 + (lane & 15);
  o[i] = (__bf16)W[k * N + n];
}

__global__ void deg_count(const int* __restrict__ idx, float* __restrict__ deg, int n) {
  int i = blockIdx.x * blockDim.x + threadIdx.x;
  if (i < n) unsafeAtomicAdd(&deg[idx[i]], 1.0f);
}

__global__ void rs_fin(float* p, int n) {  // deg -> rsqrt(clip(deg,1))
  int i = blockIdx.x * blockDim.x + threadIdx.x;
  if (i < n) p[i] = rsqrtf(fmaxf(p[i], 1.0f));
}

// --------------- edge gather/scatter (memory-bound hot loop) ---------------
// one wave per edge; lane l moves float4 l of the 128-float feature row.
__global__ void edge_scatter(const float4* __restrict__ x, const float* __restrict__ rs_src,
                             const int* __restrict__ src, const int* __restrict__ dst,
                             float* __restrict__ agg, int nedges) {
  int e = blockIdx.x * (blockDim.x >> 5) + (threadIdx.x >> 5);
  int lane = threadIdx.x & 31;
  if (e >= nedges) return;
  int s = src[e], d = dst[e];
  float sc = rs_src[s];
  float4 v = x[(size_t)s * 32 + lane];
  float* a = agg + (size_t)d * 128 + lane * 4;
  unsafeAtomicAdd(a + 0, v.x * sc);
  unsafeAtomicAdd(a + 1, v.y * sc);
  unsafeAtomicAdd(a + 2, v.z * sc);
  unsafeAtomicAdd(a + 3, v.w * sc);
}

// --------------- fused  (rs_dst * agg) @ W1 + b1 -> leaky -> @ W2 + b2 ---------------
// block = 256 threads = 8 waves; block tile = 128 rows; wave owns a 16-row stripe.
// W1f/W2f are pre-swizzled bf16 B-fragments (32B contiguous per lane, L2-resident).
__launch_bounds__(256)
__global__ void fused_gemm(const float* __restrict__ agg, const float* __restrict__ rs,
                           const __bf16* __restrict__ W1f, const float* __restrict__ b1,
                           const __bf16* __restrict__ W2f, const float* __restrict__ b2,
                           float* __restrict__ out, int nrows) {
  extern __shared__ __bf16 sA[];       // 128 x LDA (A tile, reused as H tile)
  const int tid  = threadIdx.x;
  const int row0 = blockIdx.x * 128;

  // stage A tile: normalize by rs[row], cvt to bf16, 8-elem packed stores
  for (int i = tid; i < 128 * 16; i += 256) {
    int r = i >> 4, c = (i & 15) * 8, g = row0 + r;
    float4 v0 = {0, 0, 0, 0}, v1 = {0, 0, 0, 0};
    float sc = 0.0f;
    if (g < nrows) {
      sc = rs[g];
      const float4* p = (const float4*)(agg + (size_t)g * 128 + c);
      v0 = p[0]; v1 = p[1];
    }
    Pack8 u;
    u.h[0] = (__bf16)(v0.x * sc); u.h[1] = (__bf16)(v0.y * sc);
    u.h[2] = (__bf16)(v0.z * sc); u.h[3] = (__bf16)(v0.w * sc);
    u.h[4] = (__bf16)(v1.x * sc); u.h[5] = (__bf16)(v1.y * sc);
    u.h[6] = (__bf16)(v1.z * sc); u.h[7] = (__bf16)(v1.w * sc);
    *(uint4*)(sA + r * LDA + c) = u.q;
  }
  __syncthreads();

  const int lane = tid & 31;
  const int wave = tid >> 5;
  const int half = lane >> 4;   // K-half select per ISA A/B layouts
  const int ln   = lane & 15;
  const int m0   = wave * 16;

  // ---- GEMM1: 16x128 stripe of h = A @ W1 + b1 ----
  v8f acc[8];
#pragma unroll
  for (int t = 0; t < 8; ++t) {
    float bv = b1[t * 16 + ln];
#pragma unroll
    for (int r = 0; r < 8; ++r) acc[t][r] = bv;
  }
#pragma unroll
  for (int k0 = 0; k0 < 128; k0 += 32) {
    const __bf16* arow = sA + (m0 + ln) * LDA;
    Frag af;
    af.q[0] = *(const uint4*)(arow + k0 + half * 8);        // ds_load_b128
    af.q[1] = *(const uint4*)(arow + k0 + 16 + half * 8);   // ds_load_b128
#pragma unroll
    for (int t = 0; t < 8; ++t) {
      Frag bf;
      const uint4* wq = (const uint4*)(W1f + (size_t)((((k0 >> 5) * 8 + t) * 32 + lane) << 4));
      bf.q[0] = wq[0]; bf.q[1] = wq[1];                     // global_load_b128 x2, coalesced
      acc[t] = __builtin_amdgcn_wmma_f32_16x16x32_bf16(false, af.v, false, bf.v,
                                                       (short)0, acc[t], false, false);
    }
  }
  // LeakyReLU, write H back into sA (wave-local rows -> no barrier needed)
#pragma unroll
  for (int t = 0; t < 8; ++t) {
#pragma unroll
    for (int r = 0; r < 8; ++r) {
      float v = acc[t][r];
      v = v > 0.0f ? v : 0.01f * v;
      sA[(m0 + half * 8 + r) * LDA + t * 16 + ln] = (__bf16)v;
    }
  }

  // ---- GEMM2: out = H @ W2 + b2 ----
  v8f acc2[4];
#pragma unroll
  for (int t = 0; t < 4; ++t) {
    float bv = b2[t * 16 + ln];
#pragma unroll
    for (int r = 0; r < 8; ++r) acc2[t][r] = bv;
  }
#pragma unroll
  for (int k0 = 0; k0 < 128; k0 += 32) {
    const __bf16* arow = sA + (m0 + ln) * LDA;
    Frag af;
    af.q[0] = *(const uint4*)(arow + k0 + half * 8);
    af.q[1] = *(const uint4*)(arow + k0 + 16 + half * 8);
#pragma unroll
    for (int t = 0; t < 4; ++t) {
      Frag bf;
      const uint4* wq = (const uint4*)(W2f + (size_t)((((k0 >> 5) * 4 + t) * 32 + lane) << 4));
      bf.q[0] = wq[0]; bf.q[1] = wq[1];
      acc2[t] = __builtin_amdgcn_wmma_f32_16x16x32_bf16(false, af.v, false, bf.v,
                                                        (short)0, acc2[t], false, false);
    }
  }
#pragma unroll
  for (int t = 0; t < 4; ++t) {
#pragma unroll
    for (int r = 0; r < 8; ++r) {
      int g = row0 + m0 + half * 8 + r;
      if (g < nrows) out[(size_t)g * 64 + t * 16 + ln] = acc2[t][r];
    }
  }
}

// ---------------- host-side orchestration ----------------

extern "C" void kernel_launch(void* const* d_in, const int* in_sizes, int n_in,
                              void* d_out, int out_size, void* d_ws, size_t ws_size,
                              hipStream_t stream) {
  const int F = 128, HID = 128, OUT = 64;
  const float* x_svc   = (const float*)d_in[0];
  const float* x_pod   = (const float*)d_in[1];
  const float* x_node  = (const float*)d_in[2];
  const int* svc_src   = (const int*)d_in[3];
  const int* svc_dst   = (const int*)d_in[4];
  const int* pn_src    = (const int*)d_in[5];
  const int* pn_dst    = (const int*)d_in[6];
  const int* np_src    = (const int*)d_in[7];
  const int* np_dst    = (const int*)d_in[8];
  const float* W_call  = (const float*)d_in[9];
  const float* b_call  = (const float*)d_in[10];
  const float* W_in    = (const float*)d_in[11];
  const float* b_in    = (const float*)d_in[12];
  const float* W_ni    = (const float*)d_in[13];
  const float* b_ni    = (const float*)d_in[14];
  const float* W_l_svc  = (const float*)d_in[15];
  const float* b_l_svc  = (const float*)d_in[16];
  const float* W_l_node = (const float*)d_in[17];
  const float* b_l_node = (const float*)d_in[18];
  const float* W_l_pod  = (const float*)d_in[19];
  const float* b_l_pod  = (const float*)d_in[20];

  const int nsvc  = in_sizes[0] / F;
  const int npod  = in_sizes[1] / F;
  const int nnode = in_sizes[2] / F;
  const int e_svc = in_sizes[3], e_pn = in_sizes[5], e_np = in_sizes[7];

  // workspace layout (float region, then 64B-aligned bf16 weight region)
  float* agg_svc  = (float*)d_ws;
  float* agg_node = agg_svc  + (size_t)nsvc  * F;
  float* agg_pod  = agg_node + (size_t)nnode * F;
  float* deg_svc_src = agg_pod + (size_t)npod * F;
  float* deg_svc_dst = deg_svc_src + nsvc;
  float* deg_pn_src  = deg_svc_dst + nsvc;
  float* deg_pn_dst  = deg_pn_src  + npod;
  float* deg_np_src  = deg_pn_dst  + nnode;
  float* deg_np_dst  = deg_np_src  + nnode;
  size_t woff = (size_t)((char*)(deg_np_dst + npod) - (char*)d_ws);
  woff = (woff + 63) & ~(size_t)63;
  __bf16* W1s = (__bf16*)((char*)d_ws + woff);
  __bf16* W1n = W1s + F * HID;
  __bf16* W1p = W1n + F * HID;
  __bf16* W2s = W1p + F * HID;
  __bf16* W2n = W2s + HID * OUT;
  __bf16* W2p = W2n + HID * OUT;

  // 1) zero agg + degree buffers (must happen every call)
  long nzero = (long)(nsvc + nnode + npod) * F + 2L * (nsvc + npod + nnode);
  zero_f<<<2048, 256, 0, stream>>>((float*)d_ws, nzero);

  // 2) swizzle weights into bf16 B-fragment layout
  swz_b<<<(F * HID + 255) / 256, 256, 0, stream>>>(W_call,   W1s, HID);
  swz_b<<<(F * HID + 255) / 256, 256, 0, stream>>>(W_in,     W1n, HID);
  swz_b<<<(F * HID + 255) / 256, 256, 0, stream>>>(W_ni,     W1p, HID);
  swz_b<<<(HID * OUT + 255) / 256, 256, 0, stream>>>(W_l_svc,  W2s, OUT);
  swz_b<<<(HID * OUT + 255) / 256, 256, 0, stream>>>(W_l_node, W2n, OUT);
  swz_b<<<(HID * OUT + 255) / 256, 256, 0, stream>>>(W_l_pod,  W2p, OUT);

  // 3) degrees -> rsqrt factors
  deg_count<<<(e_svc + 255) / 256, 256, 0, stream>>>(svc_src, deg_svc_src, e_svc);
  deg_count<<<(e_svc + 255) / 256, 256, 0, stream>>>(svc_dst, deg_svc_dst, e_svc);
  deg_count<<<(e_pn  + 255) / 256, 256, 0, stream>>>(pn_src,  deg_pn_src,  e_pn);
  deg_count<<<(e_pn  + 255) / 256, 256, 0, stream>>>(pn_dst,  deg_pn_dst,  e_pn);
  deg_count<<<(e_np  + 255) / 256, 256, 0, stream>>>(np_src,  deg_np_src,  e_np);
  deg_count<<<(e_np  + 255) / 256, 256, 0, stream>>>(np_dst,  deg_np_dst,  e_np);
  int ndeg = 2 * (nsvc + npod + nnode);
  rs_fin<<<(ndeg + 255) / 256, 256, 0, stream>>>(deg_svc_src, ndeg);

  // 4) edge gather/scatter (one wave per edge)
  edge_scatter<<<(e_svc + 7) / 8, 256, 0, stream>>>((const float4*)x_svc,  deg_svc_src,
                                                    svc_src, svc_dst, agg_svc, e_svc);
  edge_scatter<<<(e_pn  + 7) / 8, 256, 0, stream>>>((const float4*)x_pod,  deg_pn_src,
                                                    pn_src, pn_dst, agg_node, e_pn);
  edge_scatter<<<(e_np  + 7) / 8, 256, 0, stream>>>((const float4*)x_node, deg_np_src,
                                                    np_src, np_dst, agg_pod, e_np);

  // 5) fused dst-norm + GEMM1 + LeakyReLU + GEMM2 (WMMA bf16)
  float* out = (float*)d_out;
  size_t lds = (size_t)(128 * LDA) * sizeof(__bf16);
  fused_gemm<<<(nsvc + 127) / 128, 256, lds, stream>>>(agg_svc, deg_svc_dst, W1s, b_call,
                                                       W2s, b_l_svc, out, nsvc);
  fused_gemm<<<(nnode + 127) / 128, 256, lds, stream>>>(agg_node, deg_pn_dst, W1n, b_in,
                                                        W2n, b_l_node,
                                                        out + (size_t)nsvc * OUT, nnode);
  fused_gemm<<<(npod + 127) / 128, 256, lds, stream>>>(agg_pod, deg_np_dst, W1p, b_ni,
                                                       W2p, b_l_pod,
                                                       out + (size_t)(nsvc + nnode) * OUT, npod);
}
